// GPT_66760971649690
// MI455X (gfx1250) — matrix operand order repaired and checked by reading, not compiled, not run on hardware
//
#include <hip/hip_runtime.h>
#include <math.h>

typedef __bf16 bf16;
typedef __attribute__((ext_vector_type(8)))  __bf16 v8bf;
typedef __attribute__((ext_vector_type(16))) __bf16 v16bf;
typedef __attribute__((ext_vector_type(8)))  float  v8f;

#define WMMA_BF16(a, b, c) \
  __builtin_amdgcn_wmma_f32_16x16x32_bf16(false, (a), false, (b), (short)0, (c), false, false)

// ---------------------------------------------------------------- helpers
__device__ __forceinline__ bf16 f2bf(float f) {
  union { float f; unsigned u; } a; a.f = f;
  unsigned r = a.u + 0x7FFFu + ((a.u >> 16) & 1u);   // round-to-nearest-even
  union { unsigned short s; bf16 b; } c; c.s = (unsigned short)(r >> 16);
  return c.b;
}

__device__ __forceinline__ v16bf cat(v8bf lo, v8bf hi) {
  return __builtin_shufflevector(lo, hi, 0, 1, 2, 3, 4, 5, 6, 7,
                                         8, 9, 10, 11, 12, 13, 14, 15);
}

// ---------------------------------------------------------------- weight convert
// in: fp32 [K, N] row-major  ->  out: bf16 [N, K] (transposed, K-major rows)
__global__ void cvt_t_kernel(const float* __restrict__ in, bf16* __restrict__ out,
                             int K, int N) {
  long idx = (long)blockIdx.x * blockDim.x + threadIdx.x;
  long total = (long)K * N;
  if (idx >= total) return;
  int n = (int)(idx % N);
  int k = (int)(idx / N);
  out[(long)n * K + k] = f2bf(in[idx]);
}

// plain fp32 -> bf16 (head_W is already [N][K])
__global__ void cvt_kernel(const float* __restrict__ in, bf16* __restrict__ out, long n) {
  long idx = (long)blockIdx.x * blockDim.x + threadIdx.x;
  if (idx < n) out[idx] = f2bf(in[idx]);
}

// ---------------------------------------------------------------- embedding + pos-enc
__global__ void embed_kernel(const int* __restrict__ tokens, const float* __restrict__ emb,
                             float* __restrict__ x) {
  int row = blockIdx.x;            // b*S + s, 0..4095
  int s   = row & 1023;
  int tok = tokens[row];
  const float* e  = emb + (long)tok * 768;
  float*       xr = x + (long)row * 768;
  const float c = 9.210340371976184f / 768.0f;   // ln(10000)/D
  for (int i = threadIdx.x; i < 768; i += blockDim.x) {
    int pair  = i >> 1;
    float div = __expf(-(float)(2 * pair) * c);
    float ang = (float)s * div;
    float pe  = (i & 1) ? __cosf(ang) : __sinf(ang);
    xr[i] = e[i] + pe;
  }
}

// ---------------------------------------------------------------- layernorm (fp32 -> bf16)
__global__ __launch_bounds__(256)
void ln_kernel(const float* __restrict__ x, const float* __restrict__ g,
               const float* __restrict__ b, bf16* __restrict__ out) {
  int row = blockIdx.x;
  const float* xr = x + (long)row * 768;
  __shared__ float sbuf[256], s2buf[256];
  float s = 0.f, s2 = 0.f;
  for (int i = threadIdx.x; i < 768; i += blockDim.x) {
    float v = xr[i]; s += v; s2 += v * v;
  }
  sbuf[threadIdx.x] = s; s2buf[threadIdx.x] = s2;
  __syncthreads();
  for (int off = 128; off > 0; off >>= 1) {
    if (threadIdx.x < off) {
      sbuf[threadIdx.x]  += sbuf[threadIdx.x + off];
      s2buf[threadIdx.x] += s2buf[threadIdx.x + off];
    }
    __syncthreads();
  }
  float mu   = sbuf[0] * (1.0f / 768.0f);
  float var  = s2buf[0] * (1.0f / 768.0f) - mu * mu;
  float rstd = rsqrtf(var + 1e-5f);
  bf16* orow = out + (long)row * 768;
  for (int i = threadIdx.x; i < 768; i += blockDim.x)
    orow[i] = f2bf((xr[i] - mu) * rstd * g[i] + b[i]);
}

// ---------------------------------------------------------------- WMMA GEMM
// C[M,N](+bias)(+gelu)(+residual) = A[M,K](bf16) @ Wt[N,K]^T(bf16)
// block: 256 threads = 8 waves; tile 256(M) x 64(N); wave tile 32x64; K-step 32.
// Double-buffered LDS, register-staged global loads, one barrier per K-step.
template <bool OUT_BF16, bool GELU, bool RES>
__global__ __launch_bounds__(256)
void gemm_kernel(const bf16* __restrict__ A, const bf16* __restrict__ Wt,
                 const float* __restrict__ bias, const float* __restrict__ res,
                 void* __restrict__ out_, int M, int N, int K) {
  __shared__ __align__(32) bf16 lds_a[2][256 * 32];
  __shared__ __align__(32) bf16 lds_b[2][64 * 32];
  const int tid  = threadIdx.x;
  const int lane = tid & 31, wave = tid >> 5;
  const int l15  = lane & 15, khf = lane >> 4;
  const int m0 = blockIdx.y * 256, n0 = blockIdx.x * 64;
  const int mw = wave * 32;

  v8f acc[2][4] = {};

  // staging assignments: A = one 32-elem row per thread, B = 8 elems per thread
  const bf16* aRow = A + (long)(m0 + tid) * K;
  const int   nb   = tid >> 2, kob = (tid & 3) * 8;
  const bf16* bRow = Wt + (long)(n0 + nb) * K + kob;

  v8bf ar[4], br;
  for (int t = 0; t < 4; ++t) ar[t] = *(const v8bf*)(aRow + t * 8);
  br = *(const v8bf*)(bRow);
  for (int t = 0; t < 4; ++t) *(v8bf*)(&lds_a[0][tid * 32 + t * 8]) = ar[t];
  *(v8bf*)(&lds_b[0][nb * 32 + kob]) = br;
  __syncthreads();

  int cur = 0;
  for (int k0 = 0; k0 < K; k0 += 32) {
    const bool more = (k0 + 32) < K;
    if (more) {   // register-stage next tile; overlap with WMMA below
      for (int t = 0; t < 4; ++t) ar[t] = *(const v8bf*)(aRow + k0 + 32 + t * 8);
      br = *(const v8bf*)(bRow + k0 + 32);
      __builtin_prefetch(aRow + k0 + 64, 0, 0);   // global_prefetch_b8
      __builtin_prefetch(bRow + k0 + 64, 0, 0);
    }

    // 4 B fragments (32x16, col=l15, K run [khf*16,+16)) shared by both A frags
    v16bf bfr[4];
    for (int j = 0; j < 4; ++j)
      bfr[j] = *(const v16bf*)(&lds_b[cur][(j * 16 + l15) * 32 + khf * 16]);

    for (int s = 0; s < 2; ++s) {
      const bf16* ap = &lds_a[cur][(mw + s * 16 + l15) * 32];
      v16bf af = *(const v16bf*)(ap + khf * 16);
      for (int j = 0; j < 4; ++j)
        acc[s][j] = WMMA_BF16(af, bfr[j], acc[s][j]);
    }

    if (more) {
      const int nxt = cur ^ 1;
      for (int t = 0; t < 4; ++t) *(v8bf*)(&lds_a[nxt][tid * 32 + t * 8]) = ar[t];
      *(v8bf*)(&lds_b[nxt][nb * 32 + kob]) = br;
    }
    __syncthreads();
    cur ^= 1;
  }

  // epilogue: C layout row = r + 8*khf (per VGPR r), col = l15
  for (int s = 0; s < 2; ++s)
    for (int j = 0; j < 4; ++j) {
      int col  = n0 + j * 16 + l15;
      float bv = bias ? bias[col] : 0.f;
      for (int r = 0; r < 8; ++r) {
        long row = m0 + mw + s * 16 + r + 8 * khf;
        float v  = acc[s][j][r] + bv;
        if (GELU) v = 0.5f * v * (1.0f + erff(v * 0.70710678118654752f));
        long idx = row * (long)N + col;
        if (RES) v += res[idx];
        if (OUT_BF16) ((bf16*)out_)[idx] = f2bf(v);
        else          ((float*)out_)[idx] = v;
      }
    }
}

// ---------------------------------------------------------------- flash attention
// one wave per (b, h, 16-row q tile); q/k/v/o are bf16 [B*S, 768], head h at col h*64.
__global__ __launch_bounds__(32)
void attn_kernel(const bf16* __restrict__ q, const bf16* __restrict__ k,
                 const bf16* __restrict__ v, bf16* __restrict__ o) {
  const int qt = blockIdx.x, h = blockIdx.y, b = blockIdx.z;
  const int lane = threadIdx.x;
  const int l15 = lane & 15, khf = lane >> 4;
  const int qi = qt * 16;
  const long rowbase = (long)b * 1024;
  const float scale = 0.125f;            // 1/sqrt(64)

  __shared__ __align__(32) bf16 lds_p[16 * 32];   // P tile, row-major [q][kk]
  __shared__ __align__(32) bf16 lds_v[64 * 32];   // V tile transposed [d][kk]

  // Q fragments (two K=32 halves of dk=64)
  v16bf qa[2];
  {
    const bf16* qrow = q + (rowbase + qi + l15) * 768 + h * 64;
    for (int dh = 0; dh < 2; ++dh)
      qa[dh] = *(const v16bf*)(qrow + dh * 32 + khf * 16);
  }

  v8f acc[4] = {};
  float mrow[8], lrow[8];
  for (int r = 0; r < 8; ++r) { mrow[r] = -INFINITY; lrow[r] = 0.f; }

  for (int kbase = 0; kbase < qi + 16; kbase += 32) {
    // ---- scores: two 16x16 C tiles over 32 keys
    float p[2][8], chmax[8];
    for (int r = 0; r < 8; ++r) chmax[r] = -INFINITY;
    for (int kt = 0; kt < 2; ++kt) {
      const bf16* krow = k + (rowbase + kbase + kt * 16 + l15) * 768 + h * 64;
      v8f c = {};
      for (int dh = 0; dh < 2; ++dh) {
        v16bf kb = *(const v16bf*)(krow + dh * 32 + khf * 16);
        c = WMMA_BF16(qa[dh], kb, c);
      }
      int key = kbase + kt * 16 + l15;
      for (int r = 0; r < 8; ++r) {
        int qr = qi + r + 8 * khf;
        float s = (key <= qr) ? c[r] * scale : -INFINITY;
        p[kt][r] = s;
        chmax[r] = fmaxf(chmax[r], s);
      }
    }
    // row-max across the 16-lane half holding each row
    for (int off = 1; off < 16; off <<= 1)
      for (int r = 0; r < 8; ++r)
        chmax[r] = fmaxf(chmax[r], __shfl_xor(chmax[r], off, 32));

    float corr[8];
    for (int r = 0; r < 8; ++r) {
      float nm = fmaxf(mrow[r], chmax[r]);
      corr[r]  = __expf(mrow[r] - nm);
      mrow[r]  = nm;
    }
    for (int j = 0; j < 4; ++j)
      for (int r = 0; r < 8; ++r) acc[j][r] *= corr[r];

    // exponentiate, stash P (bf16) to LDS, accumulate row sums
    float psum[8];
    for (int r = 0; r < 8; ++r) psum[r] = 0.f;
    for (int kt = 0; kt < 2; ++kt)
      for (int r = 0; r < 8; ++r) {
        float e = __expf(p[kt][r] - mrow[r]);   // exp(-inf)=0 handles mask
        psum[r] += e;
        lds_p[(r + 8 * khf) * 32 + kt * 16 + l15] = f2bf(e);
      }
    for (int off = 1; off < 16; off <<= 1)
      for (int r = 0; r < 8; ++r) psum[r] += __shfl_xor(psum[r], off, 32);
    for (int r = 0; r < 8; ++r) lrow[r] = lrow[r] * corr[r] + psum[r];

    // ---- stage V tile transposed: lds_v[d][kk], lane owns one key row
    {
      const bf16* vrow = v + (rowbase + kbase + lane) * 768 + h * 64;
      v8bf vv[8];
      for (int t = 0; t < 8; ++t) vv[t] = *(const v8bf*)(vrow + t * 8);
      __syncthreads();
      for (int t = 0; t < 8; ++t)
        for (int i = 0; i < 8; ++i)
          lds_v[(t * 8 + i) * 32 + lane] = vv[t][i];
    }
    __syncthreads();

    // P as A fragment (16x32)
    const bf16* pp = lds_p + l15 * 32;
    v16bf pa = *(const v16bf*)(pp + khf * 16);
    // O += P @ V  (4 column tiles of dk=64)
    for (int j = 0; j < 4; ++j) {
      v16bf vb = *(const v16bf*)(&lds_v[(j * 16 + l15) * 32 + khf * 16]);
      acc[j] = WMMA_BF16(pa, vb, acc[j]);
    }
    __syncthreads();
  }

  // epilogue: normalize and scatter bf16
  for (int j = 0; j < 4; ++j)
    for (int r = 0; r < 8; ++r) {
      long row = rowbase + qi + r + 8 * khf;
      o[row * 768 + h * 64 + j * 16 + l15] = f2bf(acc[j][r] / lrow[r]);
    }
}

// ---------------------------------------------------------------- launch
extern "C" void kernel_launch(void* const* d_in, const int* in_sizes, int n_in,
                              void* d_out, int out_size, void* d_ws, size_t ws_size,
                              hipStream_t stream) {
  const int Bsz = 4, S = 1024, D = 768, H = 12, L = 6, F = 3072, V = 32000;
  const long M = (long)Bsz * S;   // 4096

  const int*   tokens = (const int*)  d_in[0];
  const float* emb    = (const float*)d_in[1];
  const float* Wq     = (const float*)d_in[2];
  const float* bq     = (const float*)d_in[3];
  const float* Wk     = (const float*)d_in[4];
  const float* bk     = (const float*)d_in[5];
  const float* Wv     = (const float*)d_in[6];
  const float* bv     = (const float*)d_in[7];
  const float* Wo     = (const float*)d_in[8];
  const float* bo     = (const float*)d_in[9];
  const float* W1     = (const float*)d_in[10];
  const float* b1     = (const float*)d_in[11];
  const float* W2     = (const float*)d_in[12];
  const float* b2     = (const float*)d_in[13];
  const float* ln1_g  = (const float*)d_in[14];
  const float* ln1_b  = (const float*)d_in[15];
  const float* ln2_g  = (const float*)d_in[16];
  const float* ln2_b  = (const float*)d_in[17];
  const float* lnf_g  = (const float*)d_in[18];
  const float* lnf_b  = (const float*)d_in[19];
  const float* headW  = (const float*)d_in[20];

  // workspace carve-out (256B aligned)
  char* wsp = (char*)d_ws;
  auto alloc = [&](size_t bytes) -> char* {
    char* p = wsp;
    wsp += (bytes + 255) & ~(size_t)255;
    return p;
  };
  const size_t DD = (size_t)D * D, DF = (size_t)D * F;
  bf16* wq_t = (bf16*)alloc(L * DD * 2);
  bf16* wk_t = (bf16*)alloc(L * DD * 2);
  bf16* wv_t = (bf16*)alloc(L * DD * 2);
  bf16* wo_t = (bf16*)alloc(L * DD * 2);
  bf16* w1_t = (bf16*)alloc(L * DF * 2);   // [F][D] per layer
  bf16* w2_t = (bf16*)alloc(L * DF * 2);   // [D][F] per layer
  bf16* hw_b = (bf16*)alloc((size_t)V * D * 2);
  float* x   = (float*)alloc(M * D * 4);
  bf16* hbuf = (bf16*) alloc(M * D * 2);
  bf16* qb   = (bf16*) alloc(M * D * 2);
  bf16* kb   = (bf16*) alloc(M * D * 2);
  bf16* vb   = (bf16*) alloc(M * D * 2);
  bf16* ob   = (bf16*) alloc(M * D * 2);
  bf16* ffb  = (bf16*) alloc(M * F * 2);

  // ---- weight conversion (transpose to [N][K] bf16)
  {
    int gDD = (int)((DD + 255) / 256), gDF = (int)((DF + 255) / 256);
    for (int l = 0; l < L; ++l) {
      cvt_t_kernel<<<gDD, 256, 0, stream>>>(Wq + l * DD, wq_t + l * DD, D, D);
      cvt_t_kernel<<<gDD, 256, 0, stream>>>(Wk + l * DD, wk_t + l * DD, D, D);
      cvt_t_kernel<<<gDD, 256, 0, stream>>>(Wv + l * DD, wv_t + l * DD, D, D);
      cvt_t_kernel<<<gDD, 256, 0, stream>>>(Wo + l * DD, wo_t + l * DD, D, D);
      cvt_t_kernel<<<gDF, 256, 0, stream>>>(W1 + l * DF, w1_t + l * DF, D, F);
      cvt_t_kernel<<<gDF, 256, 0, stream>>>(W2 + l * DF, w2_t + l * DF, F, D);
    }
    long hn = (long)V * D;
    cvt_kernel<<<(int)((hn + 255) / 256), 256, 0, stream>>>(headW, hw_b, hn);
  }

  // ---- embedding + positional encoding
  embed_kernel<<<(int)M, 256, 0, stream>>>(tokens, emb, x);

  dim3 gD((unsigned)(D / 64), (unsigned)(M / 256));
  dim3 gF((unsigned)(F / 64), (unsigned)(M / 256));
  dim3 gV((unsigned)(V / 64), (unsigned)(M / 256));
  dim3 gA((unsigned)(S / 16), (unsigned)H, (unsigned)Bsz);

  for (int l = 0; l < L; ++l) {
    ln_kernel<<<(int)M, 256, 0, stream>>>(x, ln1_g + l * D, ln1_b + l * D, hbuf);
    gemm_kernel<true, false, false><<<gD, 256, 0, stream>>>(
        hbuf, wq_t + l * DD, bq + l * D, nullptr, qb, (int)M, D, D);
    gemm_kernel<true, false, false><<<gD, 256, 0, stream>>>(
        hbuf, wk_t + l * DD, bk + l * D, nullptr, kb, (int)M, D, D);
    gemm_kernel<true, false, false><<<gD, 256, 0, stream>>>(
        hbuf, wv_t + l * DD, bv + l * D, nullptr, vb, (int)M, D, D);
    attn_kernel<<<gA, 32, 0, stream>>>(qb, kb, vb, ob);
    gemm_kernel<false, false, true><<<gD, 256, 0, stream>>>(
        ob, wo_t + l * DD, bo + l * D, x, x, (int)M, D, D);
    ln_kernel<<<(int)M, 256, 0, stream>>>(x, ln2_g + l * D, ln2_b + l * D, hbuf);
    gemm_kernel<true, true, false><<<gF, 256, 0, stream>>>(
        hbuf, w1_t + l * DF, b1 + l * F, nullptr, ffb, (int)M, F, D);
    gemm_kernel<false, false, true><<<gD, 256, 0, stream>>>(
        ffb, w2_t + l * DF, b2 + l * D, x, x, (int)M, D, F);
  }

  // final LN + LM head (fp32 logits straight to d_out)
  ln_kernel<<<(int)M, 256, 0, stream>>>(x, lnf_g, lnf_b, hbuf);
  gemm_kernel<false, false, false><<<gV, 256, 0, stream>>>(
      hbuf, hw_b, nullptr, nullptr, (float*)d_out, (int)M, V, D);
}
